// WorldModelStep_80590766343025
// MI455X (gfx1250) — matrix operand order repaired and checked by reading, not compiled, not run on hardware
//
#include <hip/hip_runtime.h>
#include <math.h>

typedef _Float16 half_t;
typedef __attribute__((ext_vector_type(16))) _Float16 v16h;
typedef __attribute__((ext_vector_type(8)))  _Float16 v8h;
typedef __attribute__((ext_vector_type(8)))  float    v8f;

#define B_    8
#define NTOK  2048
#define DDIM  16
#define KH    8
#define HEADS 4
#define HD    8

// ---------------------------------------------------------------------------
// Kernel 1: Q/K projections  state[B,N,16] @ W[16,32] + b  ->  f16 [B,N,32]
// ---------------------------------------------------------------------------
__global__ __launch_bounds__(256) void proj_kernel(
    const float* __restrict__ state,
    const float* __restrict__ Wq, const float* __restrict__ bq,
    const float* __restrict__ Wk, const float* __restrict__ bk,
    half_t* __restrict__ Qh, half_t* __restrict__ Kh)
{
    __shared__ float sWq[16 * 32], sWk[16 * 32], sbq[32], sbk[32];
    int tid = threadIdx.x;
    for (int i = tid; i < 512; i += 256) { sWq[i] = Wq[i]; sWk[i] = Wk[i]; }
    if (tid < 32) { sbq[tid] = bq[tid]; sbk[tid] = bk[tid]; }
    __syncthreads();

    size_t n = (size_t)blockIdx.x * 256 + tid;   // flat b*N+n, total B*N = 16384
    float s[16];
#pragma unroll
    for (int d = 0; d < 16; d++) s[d] = state[n * 16 + d];
#pragma unroll
    for (int j = 0; j < 32; j++) {
        float q = sbq[j], k = sbk[j];
#pragma unroll
        for (int d = 0; d < 16; d++) {
            q += s[d] * sWq[d * 32 + j];
            k += s[d] * sWk[d * 32 + j];
        }
        Qh[n * 32 + j] = (half_t)q;
        Kh[n * 32 + j] = (half_t)k;
    }
}

// ---------------------------------------------------------------------------
// Kernel 2: flash-style attention, per-head softmax, head-averaged P@V.
// One wave = one 16-query tile; 4 waves/block share LDS tiles.
// 32 keys per iteration, double-buffered through registers:
//   S0_h = A(K rows 0..15)  x Bq[h]   S1_h = A(K rows 16..31) x Bq[h]
//   (Bq[h] = Q^T, only head h's 8 feature-rows nonzero; pre-scaled by
//    log2(e)/sqrt(hd) so softmax exp is a bare v_exp_f32 = exp2)
//   C-layout of S == A-layout of P: sub-tile 0 fills Ap[0..7] (K=8g+j),
//   sub-tile 1 fills Ap[8..15] (K=16+8g+j) -> ONE fully-utilized
//   acc_h += Ap x Bs(state, 32 key rows, all lanes live).
// No running max (scores ~N(0,1): exp f32-safe; same math as reference);
// l accumulates in-lane from the f16-rounded p the WMMA actually sums.
// Mask handling: wave-uniform branch; all-true tiles pay zero mask cost.
// ---------------------------------------------------------------------------
template<bool MASKED>
__device__ __forceinline__ void attn_heads(
    const v16h& A0, const v16h& A1, const v16h* Bq, const v16h& Bs,
    v8f* acc, float* lsum, unsigned mb, int g)
{
#pragma unroll
    for (int h = 0; h < HEADS; ++h) {
        v8f cz = {};
        v8f S0 = __builtin_amdgcn_wmma_f32_16x16x32_f16(
            false, A0, false, Bq[h], (short)0, cz, false, false);
        v8f S1 = __builtin_amdgcn_wmma_f32_16x16x32_f16(
            false, A1, false, Bq[h], (short)0, cz, false, false);
        v16h Ap;
        float part = 0.f;
#pragma unroll
        for (int r = 0; r < 8; r++) {
            float pv = __builtin_amdgcn_exp2f(S0[r]);
            if (MASKED && !((mb >> (r + 8 * g)) & 1u)) pv = 0.f;
            half_t ph = (half_t)pv;
            Ap[r] = ph;
            part += (float)ph;               // match what the WMMA sums (f16)
        }
#pragma unroll
        for (int r = 0; r < 8; r++) {
            float pv = __builtin_amdgcn_exp2f(S1[r]);
            if (MASKED && !((mb >> (16 + r + 8 * g)) & 1u)) pv = 0.f;
            half_t ph = (half_t)pv;
            Ap[8 + r] = ph;
            part += (float)ph;
        }
        lsum[h] += part;
        acc[h] = __builtin_amdgcn_wmma_f32_16x16x32_f16(
            false, Ap, false, Bs, (short)0, acc[h], false, false);
    }
}

__global__ __launch_bounds__(128) void attn_kernel(
    const float* __restrict__ state, const unsigned char* __restrict__ mask,
    const half_t* __restrict__ Qh, const half_t* __restrict__ Kh,
    const float* __restrict__ v_gains, float* __restrict__ attended)
{
    __shared__ __align__(16) half_t Ktile[32 * 32];   // [key 0..31][feature]
    __shared__ __align__(16) half_t st[32 * 16];      // [key 0..31][d] f16
    __shared__ unsigned mball[64];                    // packed mask, 32 keys/word

    const int tid  = threadIdx.x;
    const int wave = tid >> 5;
    const int lane = tid & 31;
    const int g    = lane >> 4;           // half-wave group (0/1)
    const int lm   = lane & 15;
    const int b    = blockIdx.x >> 5;     // 32 query chunks of 64 per batch
    const int q0   = (blockIdx.x & 31) * 64 + wave * 16;
    // 1/sqrt(hd) * log2(e): softmax exp becomes raw v_exp_f32 (exp2)
    const float scl = 0.35355339059327373f * 1.4426950408889634f;

    // one-time: pack the whole mask row into 64 dwords
    for (int t = tid; t < 64; t += 128) {
        unsigned u = 0;
        for (int i = 0; i < 32; i++)
            u |= (mask[b * NTOK + t * 32 + i] ? 1u : 0u) << i;
        mball[t] = u;
    }

    // Loop-invariant per-head B fragments: lane column = query lm,
    // reg j = feature row (8g+j|16+8g+j). Only head f>>3 gets nonzero rows. Pre-scaled.
    v16h Bq[HEADS];
    {
#pragma unroll
        for (int j = 0; j < 16; j++) {
            const int f = (j < 8) ? (8 * g + j) : (16 + 8 * g + (j - 8));
            float qv = (float)Qh[((size_t)(b * NTOK + q0 + lm)) * 32 + f] * scl;
            half_t qh = (half_t)qv;
#pragma unroll
            for (int h = 0; h < HEADS; h++)
                Bq[h][j] = ((f >> 3) == h) ? qh : (half_t)0.f;
        }
    }

    v8f acc[HEADS];
    float lsum[HEADS];
#pragma unroll
    for (int h = 0; h < HEADS; h++) {
        lsum[h] = 0.f;
#pragma unroll
        for (int r = 0; r < 8; r++) acc[h][r] = 0.f;
    }

    // staging: thread tid owns 16B of the K tile and 4 elements of the state
    // tile (same row/col split: row = tid>>2, col4 = (tid&3)*4)
    const int srow = tid >> 2, scol = (tid & 3) << 2;
    uint4  kreg;
    float4 sreg;
    {   // preload + store tile 0
        kreg = *(const uint4*)((const unsigned*)Kh + ((size_t)(b * NTOK + srow)) * 16 + scol);
        sreg = *(const float4*)(state + ((size_t)(b * NTOK + srow)) * 16 + scol);
        *(uint4*)((unsigned*)Ktile + tid * 4) = kreg;
        half_t* d = st + tid * 4;
        d[0] = (half_t)sreg.x; d[1] = (half_t)sreg.y;
        d[2] = (half_t)sreg.z; d[3] = (half_t)sreg.w;
    }

    for (int kt = 0; kt < NTOK / 32; ++kt) {
        __syncthreads();                       // tile kt visible in LDS
        const unsigned mb = mball[kt];

        // prefetch tile kt+1 into registers (overlaps with compute below)
        if (kt + 1 < NTOK / 32) {
            const int m1 = (kt + 1) * 32;
            kreg = *(const uint4*)((const unsigned*)Kh +
                                   ((size_t)(b * NTOK + m1 + srow)) * 16 + scol);
            sreg = *(const float4*)(state + ((size_t)(b * NTOK + m1 + srow)) * 16 + scol);
        }

        // A fragments = K-tile sub-rows: row = key, K = feature (8g+j|16+8g+j)
        v16h A0, A1;
        {
            v8h a0 = *(const v8h*)&Ktile[lm * 32 + 8 * g];
            v8h a1 = *(const v8h*)&Ktile[lm * 32 + 16 + 8 * g];
            v8h a2 = *(const v8h*)&Ktile[(16 + lm) * 32 + 8 * g];
            v8h a3 = *(const v8h*)&Ktile[(16 + lm) * 32 + 16 + 8 * g];
#pragma unroll
            for (int j = 0; j < 8; j++) {
                A0[j] = a0[j]; A0[8 + j] = a1[j];
                A1[j] = a2[j]; A1[8 + j] = a3[j];
            }
        }
        // B_s fragment: K row = key (reg, 8g+j|16+8g+j), N col = d (lm)
        v16h Bs;
        {
#pragma unroll
            for (int j = 0; j < 8; j++) {
                Bs[j] = st[(8 * g + j) * 16 + lm];
                Bs[8 + j] = st[(16 + 8 * g + j) * 16 + lm];
            }
        }

        if (mb == 0xffffffffu)
            attn_heads<false>(A0, A1, Bq, Bs, acc, lsum, mb, g);
        else
            attn_heads<true>(A0, A1, Bq, Bs, acc, lsum, mb, g);

        __syncthreads();                       // all reads of tile kt done
        if (kt + 1 < NTOK / 32) {              // commit prefetched tile kt+1
            *(uint4*)((unsigned*)Ktile + tid * 4) = kreg;
            half_t* d = st + tid * 4;
            d[0] = (half_t)sreg.x; d[1] = (half_t)sreg.y;
            d[2] = (half_t)sreg.z; d[3] = (half_t)sreg.w;
        }
    }

    // epilogue: combine key halves of l, broadcast 1/l per query row, store
    const float gain = v_gains[0] * 0.25f;
    float iv[HEADS];
#pragma unroll
    for (int h = 0; h < HEADS; h++) {
        float lt = lsum[h] + __shfl_xor(lsum[h], 16, 32);
        iv[h] = 1.f / fmaxf(lt, 1e-20f);     // lane lm holds 1/l(query lm)
    }
#pragma unroll
    for (int r = 0; r < 8; r++) {
        float v = 0.f;
#pragma unroll
        for (int h = 0; h < HEADS; h++)
            v += acc[h][r] * __shfl(iv[h], r + 8 * g, 32);
        attended[((size_t)(b * NTOK + q0 + r + 8 * g)) * 16 + lm] = v * gain;
    }
}

// ---------------------------------------------------------------------------
// Kernel 3: masked world summary  [B,16] and cnt [B]
// ---------------------------------------------------------------------------
__global__ __launch_bounds__(256) void world_kernel(
    const float* __restrict__ attended, const unsigned char* __restrict__ mask,
    float* __restrict__ world, float* __restrict__ cntw)
{
    int b = blockIdx.x, tid = threadIdx.x;
    int d = tid & 15, grp = tid >> 4;
    float acc = 0.f, c = 0.f;
    for (int n = grp; n < NTOK; n += 16) {
        float mm = mask[b * NTOK + n] ? 1.f : 0.f;
        acc += attended[((size_t)(b * NTOK + n)) * 16 + d] * mm;
        if (d == 0) c += mm;
    }
    __shared__ float red[256];
    __shared__ float cc[16];
    __shared__ float scnt;
    red[tid] = acc;
    if (d == 0) cc[grp] = c;
    __syncthreads();
    for (int s = 8; s >= 1; s >>= 1) {
        if (grp < s) red[grp * 16 + d] += red[(grp + s) * 16 + d];
        __syncthreads();
    }
    if (tid == 0) {
        float t = 0.f;
        for (int i = 0; i < 16; i++) t += cc[i];
        t = fmaxf(t, 1.f);
        cntw[b] = t;
        scnt = t;
    }
    __syncthreads();
    if (grp == 0) world[b * 16 + d] = red[d] / scnt;
}

// ---------------------------------------------------------------------------
// Kernel 4: FIM proxy per (b,k) — candidates recomputed, never stored
// ---------------------------------------------------------------------------
__global__ __launch_bounds__(256) void fim_kernel(
    const float* __restrict__ attended, const float* __restrict__ hyp,
    const float* __restrict__ W_act, const float* __restrict__ b_act,
    const float* __restrict__ W_hyp, const float* __restrict__ b_hyp,
    const unsigned char* __restrict__ mask, const float* __restrict__ cntw,
    float* __restrict__ fim_out)
{
    int b = blockIdx.x >> 3, k = blockIdx.x & 7, tid = threadIdx.x;
    __shared__ float sW[256], sb[16], shf[16];
    for (int i = tid; i < 256; i += 256) sW[i] = W_act[k * 256 + i];
    if (tid < 16) {
        float hf = b_hyp[tid];
        for (int c = 0; c < 4; c++) hf += hyp[(b * KH + k) * 4 + c] * W_hyp[c * 16 + tid];
        shf[tid] = hf;
        sb[tid] = b_act[k * 16 + tid];
    }
    __syncthreads();

    float acc = 0.f;
    for (int n = tid; n < NTOK; n += 256) {
        if (!mask[b * NTOK + n]) continue;
        const float* a = attended + ((size_t)(b * NTOK + n)) * 16;
        float ar[16];
#pragma unroll
        for (int d = 0; d < 16; d++) ar[d] = a[d];
        float ss = 0.f;
#pragma unroll
        for (int e = 0; e < 16; e++) {
            float v = sb[e] + shf[e];
#pragma unroll
            for (int d = 0; d < 16; d++) v += ar[d] * sW[d * 16 + e];
            v = tanhf(v);
            ss += v * v;
        }
        acc += ss * (1.f / 16.f);
    }
    __shared__ float red[256];
    red[tid] = acc;
    __syncthreads();
    for (int s = 128; s >= 1; s >>= 1) {
        if (tid < s) red[tid] += red[tid + s];
        __syncthreads();
    }
    if (tid == 0) fim_out[b * KH + k] = red[0] / cntw[b];
}

// ---------------------------------------------------------------------------
// Kernel 5: search plane, weights softmax, modulation, rotor product
// ---------------------------------------------------------------------------
__device__ inline float cl_sign(int a, int b)
{
    int s = 1;
#pragma unroll
    for (int i = 0; i < 4; i++)
        if ((b >> i) & 1) { if (__popc(a >> (i + 1)) & 1) s = -s; }
    if ((a & b) & 8) return 0.f;   // e0 (bit 3) squares to 0
    return (float)s;
}
__device__ inline float rev_sign(int a)
{
    int pc = __popc(a);
    return ((pc * (pc - 1) / 2) & 1) ? -1.f : 1.f;
}

__global__ __launch_bounds__(64) void plane_kernel(
    const float* __restrict__ hyp, const float* __restrict__ world,
    const float* __restrict__ fim,
    const float* __restrict__ W_ws, const float* __restrict__ b_ws,
    const float* __restrict__ W_s1, const float* __restrict__ b_s1,
    const float* __restrict__ W_s2, const float* __restrict__ b_s2,
    const float* __restrict__ W_lift, const float* __restrict__ b_lift,
    const float* __restrict__ rotors, const float* __restrict__ R_accum,
    float* __restrict__ nh_out, float* __restrict__ wt_out,
    float* __restrict__ Rn_out, float* __restrict__ modw)
{
    int t = threadIdx.x, b = t >> 3, k = t & 7;
    __shared__ float logits[8][8];
    __shared__ float nh[8][8][4];

    float ws4[4];
#pragma unroll
    for (int c = 0; c < 4; c++) {
        float v = b_ws[c];
        for (int d = 0; d < 16; d++) v += world[b * 16 + d] * W_ws[d * 4 + c];
        ws4[c] = v;
    }
    float fv = fim[b * KH + k];
    float feat[10];
#pragma unroll
    for (int c = 0; c < 4; c++) feat[c] = hyp[(b * KH + k) * 4 + c];
#pragma unroll
    for (int c = 0; c < 4; c++) feat[4 + c] = ws4[c];
    feat[8] = fv; feat[9] = fv;

    float o5[5];
#pragma unroll
    for (int c = 0; c < 5; c++) o5[c] = b_s2[c];
    for (int j = 0; j < 64; j++) {
        float h = b_s1[j];
#pragma unroll
        for (int i = 0; i < 10; i++) h += feat[i] * W_s1[i * 64 + j];
        h = fmaxf(h, 0.f);
#pragma unroll
        for (int c = 0; c < 5; c++) o5[c] += h * W_s2[j * 5 + c];
    }
#pragma unroll
    for (int c = 0; c < 4; c++) {
        float v = feat[c] + o5[c];
        nh[b][k][c] = v;
        nh_out[(b * KH + k) * 4 + c] = v;
    }
    logits[b][k] = o5[4];
    __syncthreads();

    if (k == 0) {
        float mx = -1e30f;
        for (int kk = 0; kk < 8; kk++) mx = fmaxf(mx, logits[b][kk]);
        float w[8], se = 0.f;
        for (int kk = 0; kk < 8; kk++) { w[kk] = __expf(logits[b][kk] - mx); se += w[kk]; }
        float agg[4] = {0.f, 0.f, 0.f, 0.f};
        for (int kk = 0; kk < 8; kk++) {
            w[kk] /= se;
            wt_out[b * KH + kk] = w[kk];
            for (int c = 0; c < 4; c++) agg[c] += w[kk] * nh[b][kk][c];
        }
        for (int d = 0; d < 16; d++) {
            float v = b_lift[d];
            for (int c = 0; c < 4; c++) v += agg[c] * W_lift[c * 16 + d];
            modw[b * 16 + d] = 1.f + tanhf(v);
        }
        // rotor accumulation: R = gp(R_t, R_accum); normalize
        float Rt[16], Ra[16], R[16];
        for (int d = 0; d < 16; d++) {
            float v = 0.f;
            for (int kk = 0; kk < 8; kk++) v += w[kk] * rotors[kk * 16 + d];
            Rt[d] = v;
            Ra[d] = R_accum[b * 16 + d];
        }
        for (int kk = 0; kk < 16; kk++) {
            float v = 0.f;
            for (int a = 0; a < 16; a++) v += cl_sign(a, a ^ kk) * Rt[a] * Ra[a ^ kk];
            R[kk] = v;
        }
        float sq = 0.f;
        for (int a = 0; a < 16; a++) sq += cl_sign(a, a) * R[a] * (R[a] * rev_sign(a));
        sq = fmaxf(fabsf(sq), 1e-6f);
        float inv = 1.f / sqrtf(sq);
        for (int d = 0; d < 16; d++) Rn_out[b * 16 + d] = R[d] * inv;
    }
}

// ---------------------------------------------------------------------------
// Kernel 6: fused candidates->select->modulate->gate->combine->RMS norm
// ---------------------------------------------------------------------------
__global__ __launch_bounds__(256) void final_kernel(
    const float* __restrict__ state, const float* __restrict__ attended,
    const float* __restrict__ hyp,
    const float* __restrict__ W_act, const float* __restrict__ b_act,
    const float* __restrict__ W_hyp, const float* __restrict__ b_hyp,
    const float* __restrict__ weights, const float* __restrict__ modw,
    const float* __restrict__ Wg1, const float* __restrict__ bg1,
    const float* __restrict__ Wg2, const float* __restrict__ bg2,
    const float* __restrict__ Wc1, const float* __restrict__ bc1,
    const float* __restrict__ Wc2, const float* __restrict__ bc2,
    const float* __restrict__ norm_scale,
    float* __restrict__ ns_out, float* __restrict__ gate_out)
{
    __shared__ float sWa[2048], sWg1[2048], sWg2[1024];
    __shared__ float sba[128], shf[128], sbg1[64];
    __shared__ float sbg2[16], sWc1[32], sbc1[16], sWc2[16], smod[16], snsc[16];
    __shared__ float sw[8];
    __shared__ float sbc2v;

    int tid = threadIdx.x;
    int b = blockIdx.x >> 3;
    for (int i = tid; i < 2048; i += 256) { sWa[i] = W_act[i]; sWg1[i] = Wg1[i]; }
    for (int i = tid; i < 1024; i += 256) sWg2[i] = Wg2[i];
    if (tid < 128) {
        int k = tid >> 4, e = tid & 15;
        sba[tid] = b_act[tid];
        float hf = b_hyp[e];
        for (int c = 0; c < 4; c++) hf += hyp[(b * KH + k) * 4 + c] * W_hyp[c * 16 + e];
        shf[tid] = hf;
    }
    if (tid < 64) sbg1[tid] = bg1[tid];
    if (tid < 32) sWc1[tid] = Wc1[tid];
    if (tid < 16) {
        sbg2[tid] = bg2[tid]; sbc1[tid] = bc1[tid]; sWc2[tid] = Wc2[tid];
        smod[tid] = modw[b * 16 + tid]; snsc[tid] = norm_scale[tid];
    }
    if (tid < 8) sw[tid] = weights[b * KH + tid];
    if (tid == 0) sbc2v = bc2[0];
    __syncthreads();

    size_t g = (size_t)blockIdx.x * 256 + tid;    // flat token index
    float s[16], a[16];
#pragma unroll
    for (int d = 0; d < 16; d++) { s[d] = state[g * 16 + d]; a[d] = attended[g * 16 + d]; }

    float nsv[16];
#pragma unroll
    for (int e = 0; e < 16; e++) nsv[e] = 0.f;
    for (int k = 0; k < 8; k++) {
        float wk = sw[k];
#pragma unroll
        for (int e = 0; e < 16; e++) {
            float v = sba[k * 16 + e] + shf[k * 16 + e];
#pragma unroll
            for (int d = 0; d < 16; d++) v += a[d] * sWa[k * 256 + d * 16 + e];
            nsv[e] += wk * tanhf(v);
        }
    }
#pragma unroll
    for (int e = 0; e < 16; e++) nsv[e] *= smod[e];

    float gi[32];
#pragma unroll
    for (int d = 0; d < 16; d++) { gi[d] = s[d]; gi[16 + d] = nsv[d]; }
    float gate[16];
#pragma unroll
    for (int e = 0; e < 16; e++) gate[e] = sbg2[e];
    for (int j = 0; j < 64; j++) {
        float h = sbg1[j];
#pragma unroll
        for (int i = 0; i < 32; i++) h += gi[i] * sWg1[i * 64 + j];
        h = fmaxf(h, 0.f);
#pragma unroll
        for (int e = 0; e < 16; e++) gate[e] += h * sWg2[j * 16 + e];
    }
#pragma unroll
    for (int e = 0; e < 16; e++) gate[e] = 1.f / (1.f + __expf(-gate[e]));

    float cgv = sbc2v;
#pragma unroll
    for (int j = 0; j < 16; j++) {
        float h = sbc1[j] + s[0] * sWc1[j] + nsv[0] * sWc1[16 + j];
        h = fmaxf(h, 0.f);
        cgv += h * sWc2[j];
    }
    float cg = 1.f / (1.f + __expf(-cgv));

    float tmp[16];
    tmp[0] = cg * nsv[0] + (1.f - cg) * s[0];
#pragma unroll
    for (int d = 1; d < 16; d++) tmp[d] = gate[d] * nsv[d] + (1.f - gate[d]) * s[d];

    float ms = 0.f;
#pragma unroll
    for (int d = 0; d < 16; d++) ms += tmp[d] * tmp[d];
    float inv = 1.f / sqrtf(ms * (1.f / 16.f) + 1e-6f);
#pragma unroll
    for (int d = 0; d < 16; d++) {
        ns_out[g * 16 + d] = tmp[d] * snsc[d] * inv;
        gate_out[g * 16 + d] = gate[d];
    }
}

// ---------------------------------------------------------------------------
extern "C" void kernel_launch(void* const* d_in, const int* in_sizes, int n_in,
                              void* d_out, int out_size, void* d_ws, size_t ws_size,
                              hipStream_t stream)
{
    const float* state  = (const float*)d_in[0];
    const float* hyp    = (const float*)d_in[1];
    const float* R_acc  = (const float*)d_in[2];
    const unsigned char* mask = (const unsigned char*)d_in[3];
    const float* Wq = (const float*)d_in[4];  const float* bq = (const float*)d_in[5];
    const float* Wk = (const float*)d_in[6];  const float* bk = (const float*)d_in[7];
    const float* v_gains = (const float*)d_in[8];
    const float* W_act = (const float*)d_in[9];  const float* b_act = (const float*)d_in[10];
    const float* W_hyp = (const float*)d_in[11]; const float* b_hyp = (const float*)d_in[12];
    const float* W_ws  = (const float*)d_in[13]; const float* b_ws  = (const float*)d_in[14];
    const float* W_s1  = (const float*)d_in[15]; const float* b_s1  = (const float*)d_in[16];
    const float* W_s2  = (const float*)d_in[17]; const float* b_s2  = (const float*)d_in[18];
    const float* W_lift = (const float*)d_in[19]; const float* b_lift = (const float*)d_in[20];
    const float* Wg1 = (const float*)d_in[21]; const float* bg1 = (const float*)d_in[22];
    const float* Wg2 = (const float*)d_in[23]; const float* bg2 = (const float*)d_in[24];
    const float* Wc1 = (const float*)d_in[25]; const float* bc1 = (const float*)d_in[26];
    const float* Wc2 = (const float*)d_in[27]; const float* bc2 = (const float*)d_in[28];
    const float* norm_scale = (const float*)d_in[29];
    const float* rotors = (const float*)d_in[30];

    // workspace layout
    char* w = (char*)d_ws;
    half_t* Qh = (half_t*)w;                                   // 1 MB
    half_t* Kh = (half_t*)(w + (size_t)B_ * NTOK * 32 * 2);    // 1 MB
    float* attended = (float*)(w + (size_t)2 * B_ * NTOK * 32 * 2);  // 2 MB
    float* world = (float*)(w + 4194304);                      // [B,16]
    float* cntw  = world + B_ * 16;                            // [B]
    float* modw  = cntw + B_;                                  // [B,16]

    // output layout (concatenated in return order)
    float* out      = (float*)d_out;
    float* ns_out   = out;                      // [B,N,16]  262144
    float* nh_out   = out + 262144;             // [B,K,4]   256
    float* Rn_out   = out + 262400;             // [B,16]    128
    float* fim_out  = out + 262528;             // [B,K]     64
    float* wt_out   = out + 262592;             // [B,K]     64
    float* gate_out = out + 262656;             // [B,N,16]  262144

    proj_kernel<<<64, 256, 0, stream>>>(state, Wq, bq, Wk, bk, Qh, Kh);
    attn_kernel<<<B_ * (NTOK / 64), 128, 0, stream>>>(state, mask, Qh, Kh, v_gains, attended);
    world_kernel<<<B_, 256, 0, stream>>>(attended, mask, world, cntw);
    fim_kernel<<<B_ * KH, 256, 0, stream>>>(attended, hyp, W_act, b_act, W_hyp, b_hyp,
                                            mask, cntw, fim_out);
    plane_kernel<<<1, 64, 0, stream>>>(hyp, world, fim_out, W_ws, b_ws, W_s1, b_s1,
                                       W_s2, b_s2, W_lift, b_lift, rotors, R_acc,
                                       nh_out, wt_out, Rn_out, modw);
    final_kernel<<<64, 256, 0, stream>>>(state, attended, hyp, W_act, b_act, W_hyp, b_hyp,
                                         wt_out, modw, Wg1, bg1, Wg2, bg2,
                                         Wc1, bc1, Wc2, bc2, norm_scale, ns_out, gate_out);
}
